// HetEmotionNet_37580963840610
// MI455X (gfx1250) — 16-bit and 8-bit matrix operand order repaired and checked by reading, compiled, not run on hardware
//
#include <hip/hip_runtime.h>


typedef __attribute__((ext_vector_type(16))) _Float16 v16h;
typedef __attribute__((ext_vector_type(8)))  float    v8f;
typedef __attribute__((ext_vector_type(8)))  int      v8i;

#define NB     64
#define NV     40
#define NT     512
#define NFB    4
#define NCOL   516
#define NH     120
#define NY     80
#define PROJ   256
#define KCAT   40960
#define NPAIR  (NV * (NV + 1) / 2)
#define TRI_BASE(ii) ((ii) * NV - ((ii) * ((ii) - 1)) / 2)

__global__ __launch_bounds__(256) void k_concat(const float* __restrict__ eeg,
                                                const float* __restrict__ bio,
                                                float* __restrict__ xall) {
  long i = (long)blockIdx.x * blockDim.x + threadIdx.x;
  long total = (long)NB * NV * NCOL;
  if (i >= total) return;
  long per = (long)NV * NCOL;
  int b = (int)(i / per);
  long rem = i % per;
  int v = (int)(rem / NCOL), t = (int)(rem % NCOL);
  xall[i] = (v < 32) ? eeg[((long)b * 32 + v) * NCOL + t]
                     : bio[((long)b * 8 + (v - 32)) * NCOL + t];
}

__global__ __launch_bounds__(128) void k_bins(const float* __restrict__ xall,
                                              unsigned char* __restrict__ bins) {
  int wid = threadIdx.x >> 5, lane = threadIdx.x & 31;
  int row = blockIdx.x * 4 + wid;
  if (row >= NB * NV) return;
  const float* src = xall + (long)row * NCOL + NFB;
  float vals[16], mn = 3.4e38f, mx = -3.4e38f;
#pragma unroll
  for (int e = 0; e < 16; ++e) {
    float v = src[lane + 32 * e];
    vals[e] = v; mn = fminf(mn, v); mx = fmaxf(mx, v);
  }
  for (int d = 1; d < 32; d <<= 1) {
    mn = fminf(mn, __shfl_xor(mn, d, 32));
    mx = fmaxf(mx, __shfl_xor(mx, d, 32));
  }
  float sc = 10.f / (mx - mn);
#pragma unroll
  for (int e = 0; e < 16; ++e) {
    int id = (int)floorf((vals[e] - mn) * sc);
    id = id < 0 ? 0 : (id > 9 ? 9 : id);
    bins[(long)row * NT + lane + 32 * e] = (unsigned char)id;
  }
}

__device__ __forceinline__ unsigned pack_eq(unsigned w, unsigned v) {
  unsigned r = 0;
  r |= (((w      ) & 0xFFu) == v) ? 0x00000001u : 0u;
  r |= (((w >>  8) & 0xFFu) == v) ? 0x00000100u : 0u;
  r |= (((w >> 16) & 0xFFu) == v) ? 0x00010000u : 0u;
  r |= (((w >> 24) & 0xFFu) == v) ? 0x01000000u : 0u;
  return r;
}

__global__ __launch_bounds__(256) void k_onehot(
    const unsigned char* __restrict__ bins, unsigned char* __restrict__ OH) {
  long i = (long)blockIdx.x * 256 + threadIdx.x;
  long total = (long)NB * NV * 16 * (NT / 4);
  if (i >= total) return;
  long t4 = i & 127;
  long rp = i >> 7;
  long row = rp >> 4;
  unsigned p = (unsigned)(rp & 15);
  unsigned w = *(const unsigned*)(bins + row * NT + t4 * 4);
  *(unsigned*)(OH + rp * NT + t4 * 4) = pack_eq(w, p);
}

__global__ __launch_bounds__(128) void k_adjacency(
    const unsigned char* __restrict__ OH, float* __restrict__ H) {
  int wid = threadIdx.x >> 5, lane = threadIdx.x & 31;
  int task = blockIdx.x * 4 + wid;
  if (task >= NB * NPAIR) return;
  int b = task / NPAIR, t = task % NPAIR;
  int i = (int)floorf((2.f * NV + 1.f -
                       sqrtf((2.f * NV + 1.f) * (2.f * NV + 1.f) - 8.f * (float)t)) * 0.5f);
  if (i < 0) i = 0;
  if (i > NV - 1) i = NV - 1;
  while (i > 0 && TRI_BASE(i) > t) --i;
  while (i < NV - 1 && TRI_BASE(i + 1) <= t) ++i;
  int j = i + (t - TRI_BASE(i));

  int half = lane >> 4, m = lane & 15;
  const unsigned char* pi = OH + (((long)b * NV + i) * 16 + m) * NT;
  const unsigned char* pj = OH + (((long)b * NV + j) * 16 + m) * NT;

  v8i acc = {0, 0, 0, 0, 0, 0, 0, 0};
  for (int t0 = 0; t0 < NT; t0 += 64) {
    v8i a, bm;
#pragma unroll
    for (int g = 0; g < 4; ++g) {
      uint2 w = *(const uint2*)(pi + t0 + half * 16 + (g >> 1) * 32 + (g & 1) * 8);
      a[2 * g] = (int)w.x;
      a[2 * g + 1] = (int)w.y;
    }
#pragma unroll
    for (int g = 0; g < 2; ++g) {
      uint4 w = *(const uint4*)(pj + t0 + half * 16 + g * 32);
      bm[4 * g] = (int)w.x;
      bm[4 * g + 1] = (int)w.y;
      bm[4 * g + 2] = (int)w.z;
      bm[4 * g + 3] = (int)w.w;
    }
    acc = __builtin_amdgcn_wmma_i32_16x16x64_iu8(false, a, false, bm, acc,
                                                 false, false);
  }

  int colsum = 0;
#pragma unroll
  for (int r = 0; r < 8; ++r) colsum += acc[r];
  int nj = colsum + __shfl_xor(colsum, 16, 32);

  float mi = 0.f;
  const float logN = __logf((float)NT);
#pragma unroll
  for (int r = 0; r < 8; ++r) {
    int ni = acc[r];
    ni += __shfl_xor(ni, 1, 32);
    ni += __shfl_xor(ni, 2, 32);
    ni += __shfl_xor(ni, 4, 32);
    ni += __shfl_xor(ni, 8, 32);
    int c = acc[r];
    if (c > 0)
      mi += ((float)c / (float)NT) *
            (__logf((float)c) - __logf((float)ni) - __logf((float)nj) + logN);
  }
  for (int d = 1; d < 32; d <<= 1) mi += __shfl_xor(mi, d, 32);
  if (lane == 0) {
    mi = fmaxf(mi, 0.f);
    if (!(mi < 3.0e38f)) mi = 0.f;
    float out = (mi < 0.7f) ? 0.f : mi;
    H[((long)b * NV + i) * NV + j] = out;
    H[((long)b * NV + j) * NV + i] = out;
  }
}

__global__ __launch_bounds__(256) void k_norm_adj(const float* __restrict__ H,
                                                  float* __restrict__ Hn) {
  int b = blockIdx.x;
  __shared__ float sHs[NV][NV];
  __shared__ float sD[NV];
  for (int i = threadIdx.x; i < NV * NV; i += 256) {
    int r = i / NV, c = i % NV;
    sHs[r][c] = (r == c) ? 1.f : H[(long)b * NV * NV + i];
  }
  __syncthreads();
  if (threadIdx.x < NV) {
    float s = 0.f;
    for (int c = 0; c < NV; ++c) s += sHs[threadIdx.x][c];
    sD[threadIdx.x] = rsqrtf(s);
  }
  __syncthreads();
  for (int i = threadIdx.x; i < NV * NV; i += 256) {
    int r = i / NV, c = i % NV;
    Hn[(long)b * NV * NV + i] = sD[r] * sHs[r][c] * sD[c];
  }
}

template <bool AT, bool BT, bool RELU>
__global__ __launch_bounds__(128) void k_gemm(
    const float* __restrict__ A, const float* __restrict__ Bm,
    const float* __restrict__ bias, const float* __restrict__ scaleptr,
    float* __restrict__ C, int M, int N, int K, int lda, int ldb, int ldc,
    long aBS, long bBS, long cBS, float slope) {
  int wid = threadIdx.x >> 5, lane = threadIdx.x & 31;
  int tnC = (N + 15) >> 4, tmC = (M + 15) >> 4;
  int tile = blockIdx.x * 4 + wid;
  if (tile >= tmC * tnC) return;
  int tm = tile / tnC, tn = tile % tnC;
  const float* Ab = A + (long)blockIdx.z * aBS;
  const float* Bb = Bm + (long)blockIdx.z * bBS;
  float* Cb = C + (long)blockIdx.z * cBS;
  int half = lane >> 4, nl = lane & 15;
  int rowA = tm * 16 + nl, colB = tn * 16 + nl;
  const bool vecA = (!AT) && ((K & 31) == 0) && ((lda & 3) == 0);
  const bool vecB = BT && ((K & 31) == 0) && ((ldb & 3) == 0);

  v8f acc = {0, 0, 0, 0, 0, 0, 0, 0};
  int ksteps = (K + 31) >> 5;
  for (int kk = 0; kk < ksteps; ++kk) {
    int k0 = kk << 5;
    if (kk + 1 < ksteps) {
      if (rowA < M) __builtin_prefetch(AT ? &Ab[(long)(k0 + 32) * lda + rowA]
                                          : &Ab[(long)rowA * lda + k0 + 32], 0, 1);
      if (colB < N) __builtin_prefetch(BT ? &Bb[(long)colB * ldb + k0 + 32]
                                          : &Bb[(long)(k0 + 32) * ldb + colB], 0, 1);
    }
    v16h af, bf;
    if (vecA) {
      if (rowA < M) {
        const float* base = Ab + (long)rowA * lda + k0 + half * 16;
        float4 a0 = *(const float4*)(base);
        float4 a1 = *(const float4*)(base + 4);
        float4 a2 = *(const float4*)(base + 8);
        float4 a3 = *(const float4*)(base + 12);
        af[0] = (_Float16)a0.x; af[1] = (_Float16)a0.y;
        af[2] = (_Float16)a0.z; af[3] = (_Float16)a0.w;
        af[4] = (_Float16)a1.x; af[5] = (_Float16)a1.y;
        af[6] = (_Float16)a1.z; af[7] = (_Float16)a1.w;
        af[8] = (_Float16)a2.x; af[9] = (_Float16)a2.y;
        af[10] = (_Float16)a2.z; af[11] = (_Float16)a2.w;
        af[12] = (_Float16)a3.x; af[13] = (_Float16)a3.y;
        af[14] = (_Float16)a3.z; af[15] = (_Float16)a3.w;
      } else {
#pragma unroll
        for (int e = 0; e < 16; ++e) af[e] = (_Float16)0.f;
      }
    } else {
#pragma unroll
      for (int e = 0; e < 16; ++e) {
        int ka = k0 + half * 16 + e;
        float va = 0.f;
        if (rowA < M && ka < K)
          va = AT ? Ab[(long)ka * lda + rowA] : Ab[(long)rowA * lda + ka];
        af[e] = (_Float16)va;
      }
    }
    if (vecB) {
      if (colB < N) {
        const float* base = Bb + (long)colB * ldb + k0 + half * 16;
        float4 b0 = *(const float4*)(base);
        float4 b1 = *(const float4*)(base + 4);
        float4 b2 = *(const float4*)(base + 8);
        float4 b3 = *(const float4*)(base + 12);
        bf[0] = (_Float16)b0.x; bf[1] = (_Float16)b0.y;
        bf[2] = (_Float16)b0.z; bf[3] = (_Float16)b0.w;
        bf[4] = (_Float16)b1.x; bf[5] = (_Float16)b1.y;
        bf[6] = (_Float16)b1.z; bf[7] = (_Float16)b1.w;
        bf[8] = (_Float16)b2.x; bf[9] = (_Float16)b2.y;
        bf[10] = (_Float16)b2.z; bf[11] = (_Float16)b2.w;
        bf[12] = (_Float16)b3.x; bf[13] = (_Float16)b3.y;
        bf[14] = (_Float16)b3.z; bf[15] = (_Float16)b3.w;
      } else {
#pragma unroll
        for (int e = 0; e < 16; ++e) bf[e] = (_Float16)0.f;
      }
    } else {
#pragma unroll
      for (int e = 0; e < 16; ++e) {
        int kb = k0 + half * 16 + e;
        float vb = 0.f;
        if (colB < N && kb < K)
          vb = BT ? Bb[(long)colB * ldb + kb] : Bb[(long)kb * ldb + colB];
        bf[e] = (_Float16)vb;
      }
    }
    acc = __builtin_amdgcn_wmma_f32_16x16x32_f16(false, af, false, bf,
                                                 (short)0, acc, false, false);
  }
  float sc = scaleptr ? scaleptr[0] : 1.f;
#pragma unroll
  for (int r = 0; r < 8; ++r) {
    int row = tm * 16 + r + 8 * half;
    int col = tn * 16 + nl;
    if (row < M && col < N) {
      float v = acc[r] * sc + (bias ? bias[col] : 0.f);
      if (RELU) v = v > 0.f ? v : v * slope;
      Cb[(long)row * ldc + col] = v;
    }
  }
}

__global__ __launch_bounds__(256) void k_gru_scan(
    const float* __restrict__ gi,
    const float* __restrict__ Whh,
    const float* __restrict__ bhh,
    float* __restrict__ Y,
    int seq, int backward, int colOff) {
  __shared__ _Float16 sWhh[NH][48];
  __shared__ _Float16 sH[NB][48];
  __shared__ float sH32[NB][NV];
  __shared__ float sGh[NB][NH];
  int tid = threadIdx.x;
  for (int i = tid; i < NH * 48; i += 256) {
    int n = i / 48, k = i % 48;
    sWhh[n][k] = (k < NV) ? (_Float16)Whh[n * NV + k] : (_Float16)0.f;
  }
  for (int i = tid; i < NB * 48; i += 256) sH[i / 48][i % 48] = (_Float16)0.f;
  for (int i = tid; i < NB * NV; i += 256) sH32[i / NV][i % NV] = 0.f;
  __syncthreads();

  int wid = tid >> 5, lane = tid & 31, half = lane >> 4, nl = lane & 15;
  int tn = wid;
  for (int s = 0; s < seq; ++s) {
    int idx = backward ? (seq - 1 - s) : s;
    for (int tm = 0; tm < 4; ++tm) {
      v8f acc = {0, 0, 0, 0, 0, 0, 0, 0};
#pragma unroll
      for (int kk = 0; kk < 2; ++kk) {
        int k0 = kk * 32;
        v16h af, bf;
#pragma unroll
        for (int e = 0; e < 16; ++e) {
          int ka = k0 + half * 16 + e;
          int row = tm * 16 + nl;
          af[e] = (ka < 48) ? sH[row][ka] : (_Float16)0.f;
          int kb = k0 + half * 16 + e;
          int col = tn * 16 + nl;
          bf[e] = (col < NH && kb < 48) ? sWhh[col][kb] : (_Float16)0.f;
        }
        acc = __builtin_amdgcn_wmma_f32_16x16x32_f16(false, af, false, bf,
                                                     (short)0, acc, false, false);
      }
#pragma unroll
      for (int r = 0; r < 8; ++r) {
        int row = tm * 16 + r + 8 * half;
        int col = tn * 16 + nl;
        if (col < NH) sGh[row][col] = acc[r];
      }
    }
    __syncthreads();
    for (int p = tid; p < NB * NV; p += 256) {
      int m = p / NV, u = p % NV;
      const float* g = gi + ((long)m * seq + idx) * NH;
      float ir = g[u], iz = g[NV + u], inn = g[2 * NV + u];
      float hr = sGh[m][u] + bhh[u];
      float hz = sGh[m][NV + u] + bhh[NV + u];
      float hn = sGh[m][2 * NV + u] + bhh[2 * NV + u];
      float r  = 1.f / (1.f + __expf(-(ir + hr)));
      float zg = 1.f / (1.f + __expf(-(iz + hz)));
      float nn = tanhf(inn + r * hn);
      float hnew = (1.f - zg) * nn + zg * sH32[m][u];
      sH32[m][u] = hnew;
      sH[m][u] = (_Float16)hnew;
      Y[((long)m * seq + idx) * NY + colOff + u] = hnew;
    }
    __syncthreads();
  }
}

__global__ __launch_bounds__(256) void k_bn(const float* __restrict__ Y,
                                            float* __restrict__ Z,
                                            const float* __restrict__ gamma,
                                            const float* __restrict__ beta,
                                            int seq) {
  int f = blockIdx.x;
  __shared__ float s1[256], s2[256];
  float a = 0.f, b = 0.f;
  for (int i = threadIdx.x; i < NB * NY; i += 256) {
    int bb = i / NY, c = i % NY;
    float v = Y[((long)bb * seq + f) * NY + c];
    a += v; b += v * v;
  }
  s1[threadIdx.x] = a; s2[threadIdx.x] = b;
  __syncthreads();
  for (int d = 128; d > 0; d >>= 1) {
    if (threadIdx.x < d) {
      s1[threadIdx.x] += s1[threadIdx.x + d];
      s2[threadIdx.x] += s2[threadIdx.x + d];
    }
    __syncthreads();
  }
  float mu = s1[0] / (float)(NB * NY);
  float var = s2[0] / (float)(NB * NY) - mu * mu;
  float inv = rsqrtf(var + 1e-5f);
  float g = gamma[f], be = beta[f];
  for (int i = threadIdx.x; i < NB * NY; i += 256) {
    int bb = i / NY, c = i % NY;
    float v = Y[((long)bb * seq + f) * NY + c];
    Z[((long)bb * NY + c) * seq + f] = (v - mu) * inv * g + be;
  }
}

extern "C" void kernel_launch(void* const* d_in, const int* in_sizes, int n_in,
                              void* d_out, int out_size, void* d_ws,
                              size_t ws_size, hipStream_t stream) {
  (void)in_sizes; (void)n_in; (void)out_size; (void)ws_size;
  const float* eeg = (const float*)d_in[0];
  const float* bio = (const float*)d_in[1];
  const float* F_Wih_f = (const float*)d_in[2];
  const float* F_Whh_f = (const float*)d_in[3];
  const float* F_bih_f = (const float*)d_in[4];
  const float* F_bhh_f = (const float*)d_in[5];
  const float* F_Wih_b = (const float*)d_in[6];
  const float* F_Whh_b = (const float*)d_in[7];
  const float* F_bih_b = (const float*)d_in[8];
  const float* F_bhh_b = (const float*)d_in[9];
  const float* F_w     = (const float*)d_in[10];
  const float* F_gamma = (const float*)d_in[11];
  const float* F_beta  = (const float*)d_in[12];
  const float* T_Wih_f = (const float*)d_in[13];
  const float* T_Whh_f = (const float*)d_in[14];
  const float* T_bih_f = (const float*)d_in[15];
  const float* T_bhh_f = (const float*)d_in[16];
  const float* T_Wih_b = (const float*)d_in[17];
  const float* T_Whh_b = (const float*)d_in[18];
  const float* T_bih_b = (const float*)d_in[19];
  const float* T_bhh_b = (const float*)d_in[20];
  const float* T_w     = (const float*)d_in[21];
  const float* T_gamma = (const float*)d_in[22];
  const float* T_beta  = (const float*)d_in[23];
  const float* WF = (const float*)d_in[24];
  const float* bF = (const float*)d_in[25];
  const float* WT = (const float*)d_in[26];
  const float* bT = (const float*)d_in[27];
  const float* W1 = (const float*)d_in[28];
  const float* b1 = (const float*)d_in[29];
  const float* W2 = (const float*)d_in[30];
  const float* b2 = (const float*)d_in[31];

  char* ws = (char*)d_ws;
  size_t off = 0;
  auto alloc = [&](size_t bytes) -> void* {
    void* p = ws + off;
    off = (off + bytes + 255) & ~(size_t)255;
    return p;
  };
  float* xall = (float*)alloc((size_t)NB * NV * NCOL * 4);
  unsigned char* bins = (unsigned char*)alloc((size_t)NB * NV * NT);
  unsigned char* OH   = (unsigned char*)alloc((size_t)NB * NV * 16 * NT);
  float* H    = (float*)alloc((size_t)NB * NV * NV * 4);
  float* Hn   = (float*)alloc((size_t)NB * NV * NV * 4);
  float* XgT  = (float*)alloc((size_t)NB * NV * NT * 4);
  float* XgF  = (float*)alloc((size_t)NB * NV * NFB * 4);
  float* giTf = (float*)alloc((size_t)NB * NT * NH * 4);
  float* giTb = (float*)alloc((size_t)NB * NT * NH * 4);
  float* giFf = (float*)alloc((size_t)NB * NFB * NH * 4);
  float* giFb = (float*)alloc((size_t)NB * NFB * NH * 4);
  float* Y_T  = (float*)alloc((size_t)NB * NT * NY * 4);
  float* Y_F  = (float*)alloc((size_t)NB * NFB * NY * 4);
  float* Z_T  = (float*)alloc((size_t)NB * NY * NT * 4);
  float* Z_F  = (float*)alloc((size_t)NB * NY * NFB * 4);
  float* OC   = (float*)alloc((size_t)NB * KCAT * 4);
  float* act  = (float*)alloc((size_t)NB * 64 * 4);

  {
    long total = (long)NB * NV * NCOL;
    k_concat<<<(int)((total + 255) / 256), 256, 0, stream>>>(eeg, bio, xall);
  }
  k_bins<<<(NB * NV) / 4, 128, 0, stream>>>(xall, bins);
  {
    long total = (long)NB * NV * 16 * (NT / 4);
    k_onehot<<<(int)((total + 255) / 256), 256, 0, stream>>>(bins, OH);
  }
  k_adjacency<<<(NB * NPAIR + 3) / 4, 128, 0, stream>>>(OH, H);
  k_norm_adj<<<NB, 256, 0, stream>>>(H, Hn);
  k_gemm<false, false, true><<<dim3(24, 1, NB), 128, 0, stream>>>(
      Hn, xall + NFB, nullptr, T_w, XgT, NV, NT, NV, NV, NCOL, NT,
      (long)NV * NV, (long)NV * NCOL, (long)NV * NT, 0.01f);
  k_gemm<false, false, true><<<dim3(1, 1, NB), 128, 0, stream>>>(
      Hn, xall, nullptr, F_w, XgF, NV, NFB, NV, NV, NCOL, NFB,
      (long)NV * NV, (long)NV * NCOL, (long)NV * NFB, 0.01f);
  k_gemm<true, true, false><<<dim3(64, 1, NB), 128, 0, stream>>>(
      XgT, T_Wih_f, T_bih_f, nullptr, giTf, NT, NH, NV, NT, NV, NH,
      (long)NV * NT, 0, (long)NT * NH, 0.f);
  k_gemm<true, true, false><<<dim3(64, 1, NB), 128, 0, stream>>>(
      XgT, T_Wih_b, T_bih_b, nullptr, giTb, NT, NH, NV, NT, NV, NH,
      (long)NV * NT, 0, (long)NT * NH, 0.f);
  k_gemm<true, true, false><<<dim3(2, 1, NB), 128, 0, stream>>>(
      XgF, F_Wih_f, F_bih_f, nullptr, giFf, NFB, NH, NV, NFB, NV, NH,
      (long)NV * NFB, 0, (long)NFB * NH, 0.f);
  k_gemm<true, true, false><<<dim3(2, 1, NB), 128, 0, stream>>>(
      XgF, F_Wih_b, F_bih_b, nullptr, giFb, NFB, NH, NV, NFB, NV, NH,
      (long)NV * NFB, 0, (long)NFB * NH, 0.f);
  k_gru_scan<<<1, 256, 0, stream>>>(giTf, T_Whh_f, T_bhh_f, Y_T, NT, 0, 0);
  k_gru_scan<<<1, 256, 0, stream>>>(giTb, T_Whh_b, T_bhh_b, Y_T, NT, 1, NV);
  k_gru_scan<<<1, 256, 0, stream>>>(giFf, F_Whh_f, F_bhh_f, Y_F, NFB, 0, 0);
  k_gru_scan<<<1, 256, 0, stream>>>(giFb, F_Whh_b, F_bhh_b, Y_F, NFB, 1, NV);
  k_bn<<<NT, 256, 0, stream>>>(Y_T, Z_T, T_gamma, T_beta, NT);
  k_bn<<<NFB, 256, 0, stream>>>(Y_F, Z_F, F_gamma, F_beta, NFB);
  k_gemm<false, true, false><<<dim3(20, 1, NB), 128, 0, stream>>>(
      Z_F, WF, bF, nullptr, OC, NY, PROJ, NFB, NFB, NFB, PROJ,
      (long)NY * NFB, 0, (long)KCAT, 0.f);
  k_gemm<false, true, false><<<dim3(20, 1, NB), 128, 0, stream>>>(
      Z_T, WT, bT, nullptr, OC + NY * PROJ, NY, PROJ, NT, NT, NT, PROJ,
      (long)NY * NT, 0, (long)KCAT, 0.f);
  k_gemm<false, true, true><<<dim3(4, 1, 1), 128, 0, stream>>>(
      OC, W1, b1, nullptr, act, NB, 64, KCAT, KCAT, KCAT, 64, 0, 0, 0, 0.01f);
  k_gemm<false, true, false><<<dim3(1, 1, 1), 128, 0, stream>>>(
      act, W2, b2, nullptr, (float*)d_out, NB, 3, 64, 64, 64, 3, 0, 0, 0, 0.f);
}
